// VQVAE_35854386987356
// MI455X (gfx1250) — matrix operand order repaired and checked by reading, not compiled, not run on hardware
//
#include <hip/hip_runtime.h>
#include <math.h>

typedef __attribute__((ext_vector_type(16))) __bf16 v16bf;
typedef __attribute__((ext_vector_type(8)))  float  v8f;

// ---------------------------------------------------------------------------
// Zero the two scalar loss slots (re-run every call so atomics start clean).
// ---------------------------------------------------------------------------
__global__ void init_losses(float* __restrict__ loss2) {
    if (threadIdx.x == 0) { loss2[0] = 0.0f; loss2[1] = 0.0f; }
}

// ---------------------------------------------------------------------------
// Zero a buffer (16B granules) -- used to pre-zero the padded halos.
// ---------------------------------------------------------------------------
__global__ void __launch_bounds__(256) zero_buf(uint4* __restrict__ p, int n4) {
    int i = blockIdx.x * 256 + threadIdx.x;
    if (i < n4) { uint4 z = {0u, 0u, 0u, 0u}; p[i] = z; }
}

// ---------------------------------------------------------------------------
// Pack conv weights into fragment-ready bf16 WMMA B layout:
//   buffer[((kb32*NT + nt)*32 + lane)*16 + e] = W[k][n]
//     k = kb32*32 + (lane>>4)*16 + e,  n = nt*16 + (lane&15)
// so each wave's 32x16 B fragment is ONE contiguous v16bf (32B) per lane.
//   w2  [64][32][4][4]           -> w2p  (K=512, NT=4)
//   w3  [64][64][3][3]           -> w3p  (K=576, NT=4)
//   dw1 [in=64][out=64][3][3] (ConvTranspose) -> dw1p transposed + flipped
//   dw2 [in=64][out=32][4][4] (ConvTranspose, stride 2) -> dw2p: 4 parity
//       classes, each K=256 (k = ci*4 + th*2 + tw), N=32, NT=2.
//       NOTE: tw is address-flipped (kw uses 1-tw) so that consecutive k in a
//       pair correspond to ASCENDING input addresses (merges to b32, no swap).
// ---------------------------------------------------------------------------
__global__ void pack_weights(const float* __restrict__ w2,
                             const float* __restrict__ w3,
                             const float* __restrict__ dw1,
                             const float* __restrict__ dw2,
                             __bf16* __restrict__ w2p,
                             __bf16* __restrict__ w3p,
                             __bf16* __restrict__ dw1p,
                             __bf16* __restrict__ dw2p) {
    int i = blockIdx.x * blockDim.x + threadIdx.x;
    {   // w2p / w3p / dw1p  (NT=4)
        int e    = i & 15;
        int lane = (i >> 4) & 31;
        int nt   = (i >> 9) & 3;
        int kb32 = i >> 11;
        int k = kb32 * 32 + (lane >> 4) * 16 + e;
        int n = nt * 16 + (lane & 15);
        if (i < 32768) w2p[i] = (__bf16)w2[n * 512 + k];
        if (i < 36864) {
            w3p[i] = (__bf16)w3[n * 576 + k];
            int ci = k / 9, t = k % 9, kh = t / 3, kw = t % 3;
            dw1p[i] = (__bf16)dw1[ci * 576 + n * 9 + (2 - kh) * 3 + (2 - kw)];
        }
    }
    if (i < 32768) {  // dw2p: [cls][kb32 0..7][nt 0..1][lane][e]  (NT=2)
        int e    = i & 15;
        int lane = (i >> 4) & 31;
        int nt   = (i >> 9) & 1;
        int kb32 = (i >> 10) & 7;
        int cls  = i >> 13;
        int py = cls >> 1, px = cls & 1;
        int k = kb32 * 32 + (lane >> 4) * 16 + e;      // k = ci*4 + th*2 + tw
        int n = nt * 16 + (lane & 15);                 // out channel (0..31)
        int ci = k >> 2, t = k & 3, th = t >> 1, tw = t & 1;
        int kh = (1 - py) + 2 * th;
        int kw = (1 - px) + 2 * (1 - tw);              // tw address-flipped
        dw2p[i] = (__bf16)dw2[ci * 512 + n * 16 + kh * 4 + kw];
    }
}

// ---------------------------------------------------------------------------
// conv1: 1->32, 4x4, stride 2, pad 1, ReLU. Direct (K_gemm=16, VALU is fine).
// Writes interior of padded h1p bf16 [128][32][66][66].
// ---------------------------------------------------------------------------
__global__ void __launch_bounds__(256) conv1_relu(
        const float* __restrict__ x, const float* __restrict__ w1,
        const float* __restrict__ b1, __bf16* __restrict__ h1p) {
    int idx = blockIdx.x * 256 + threadIdx.x;          // ((b*32+o)*64+oy)*64+ox
    int ox = idx & 63;
    int oy = (idx >> 6) & 63;
    int o  = (idx >> 12) & 31;
    int b  = idx >> 17;
    const float* xb = x + (size_t)b * 16384;           // 128*128
    float acc = b1[o];
#pragma unroll
    for (int kh = 0; kh < 4; ++kh) {
        int iy = oy * 2 - 1 + kh;
        if ((unsigned)iy >= 128u) continue;
#pragma unroll
        for (int kw = 0; kw < 4; ++kw) {
            int ix = ox * 2 - 1 + kw;
            if ((unsigned)ix >= 128u) continue;
            acc += w1[o * 16 + kh * 4 + kw] * xb[iy * 128 + ix];
        }
    }
    size_t oidx = (((size_t)b * 32 + o) * 66 + oy + 1) * 66 + ox + 1;
    h1p[oidx] = (__bf16)(acc > 0.0f ? acc : 0.0f);
}

// ---------------------------------------------------------------------------
// Implicit-GEMM conv on the WMMA path (v_wmma_f32_16x16x32_bf16).
//   M = 128*32*32 = 131072 output pixels, N = 64 for all three layers.
//   Block = 128 threads = 4 waves; wave owns a 16(M) x 64(N) tile.
//   Input is zero-PADDED [CIN][HP][HP] (HP = HIN+2, pad=1) so every im2col
//   tap is an unconditional load at a constant offset.
//   OUTMODE: 0 = f32 flat [B,64,32,32], 1 = bf16 flat, 2 = bf16 padded 34x34.
// ---------------------------------------------------------------------------
template<int CIN, int KS, int STRIDE, int HP, bool RELU, int OUTMODE>
__global__ void __launch_bounds__(128) conv_gemm(
        const __bf16* __restrict__ in, const __bf16* __restrict__ wfrag,
        const float* __restrict__ bias, void* __restrict__ outp) {
    constexpr int K  = CIN * KS * KS;
    constexpr int KK = KS * KS;
    static_assert(K % 32 == 0, "K must be multiple of 32");

    const int lane  = threadIdx.x & 31;
    const int wave  = threadIdx.x >> 5;
    const int mBase = blockIdx.x * 64 + wave * 16;
    const int lhalf = lane >> 4;     // which 16-lane half
    const int lidx  = lane & 15;

    const int m   = mBase + lidx;
    const int b   = m >> 10;         // 32*32 pixels per image
    const int rem = m & 1023;
    const int oy  = rem >> 5;
    const int ox  = rem & 31;
    const __bf16* inB = in + ((size_t)b * CIN) * (HP * HP)
                           + (oy * STRIDE) * HP + (ox * STRIDE);
    const v16bf* wf = (const v16bf*)wfrag;

    const v8f zero8 = {0.f, 0.f, 0.f, 0.f, 0.f, 0.f, 0.f, 0.f};
    v8f acc[4] = {zero8, zero8, zero8, zero8};

#pragma unroll
    for (int kb = 0; kb < K; kb += 32) {
        if (kb + 32 < K) __builtin_prefetch(&wfrag[(kb + 32) * 64], 0, 1);

        v16bf a;
#pragma unroll
        for (int v = 0; v < 8; ++v) {
#pragma unroll
            for (int j = 0; j < 2; ++j) {
                int k  = kb + lhalf * 16 + v * 2 + j;
                int ci = k / KK;
                int t  = k % KK;
                int kh = t / KS;
                int kw = t % KS;
                a[2 * v + j] = inB[(ci * HP + kh) * HP + kw];
            }
        }

        const int fb = (kb >> 5) * 4;
        v16bf bf0 = wf[((fb + 0) << 5) + lane];
        v16bf bf1 = wf[((fb + 1) << 5) + lane];
        v16bf bf2 = wf[((fb + 2) << 5) + lane];
        v16bf bf3 = wf[((fb + 3) << 5) + lane];

        acc[0] = __builtin_amdgcn_wmma_f32_16x16x32_bf16(false, a, false, bf0, (short)0, acc[0], false, false);
        acc[1] = __builtin_amdgcn_wmma_f32_16x16x32_bf16(false, a, false, bf1, (short)0, acc[1], false, false);
        acc[2] = __builtin_amdgcn_wmma_f32_16x16x32_bf16(false, a, false, bf2, (short)0, acc[2], false, false);
        acc[3] = __builtin_amdgcn_wmma_f32_16x16x32_bf16(false, a, false, bf3, (short)0, acc[3], false, false);
    }

#pragma unroll
    for (int nt = 0; nt < 4; ++nt) {
        int n = nt * 16 + lidx;
        float bv = bias[n];
#pragma unroll
        for (int r = 0; r < 8; ++r) {
            int mm  = mBase + lhalf * 8 + r;
            int bb  = mm >> 10;
            int pix = mm & 1023;
            float v = acc[nt][r] + bv;
            if (RELU) v = fmaxf(v, 0.0f);
            if (OUTMODE == 0) {
                ((float*)outp)[(((size_t)bb * 64 + n) << 10) + pix] = v;
            } else if (OUTMODE == 1) {
                ((__bf16*)outp)[(((size_t)bb * 64 + n) << 10) + pix] = (__bf16)v;
            } else {
                int yy = pix >> 5, xx = pix & 31;
                ((__bf16*)outp)[(((size_t)bb * 64 + n) * 34 + yy + 1) * 34 + xx + 1] = (__bf16)v;
            }
        }
    }
}

// ---------------------------------------------------------------------------
// deconv2 on the WMMA path: ConvTranspose2d 64->32, 4x4, stride 2, pad 1.
// Decomposed into 4 output-parity classes (blockIdx.y); each class is a dense
// GEMM: M=131072 (b,qy,qx), N=32, K=256 (k = ci*4 + th*2 + tw), with taps
//   kh=(1-py)+2*th, kw=(1-px)+2*(1-tw)  (tw flipped for ascending addresses),
//   padded input row = (qy+1+py)-th, col = (qx+px)-(1-tw)+1 = qx+px+tw.
// in h3p bf16 [128][64][34][34] -> h4 bf16 [128,32,64,64], ReLU.
// ---------------------------------------------------------------------------
__global__ void __launch_bounds__(128) deconv2_gemm(
        const __bf16* __restrict__ h3p, const __bf16* __restrict__ wfrag,
        const float* __restrict__ bias, __bf16* __restrict__ h4) {
    const int cls = blockIdx.y;            // 0..3
    const int py = cls >> 1, px = cls & 1;
    const int lane  = threadIdx.x & 31;
    const int wave  = threadIdx.x >> 5;
    const int mBase = blockIdx.x * 64 + wave * 16;
    const int lhalf = lane >> 4;
    const int lidx  = lane & 15;

    const int m   = mBase + lidx;
    const int b   = m >> 10;
    const int rem = m & 1023;
    const int qy  = rem >> 5;
    const int qx  = rem & 31;
    // base at (row = qy+1+py, col = qx+px); tap offset = -th*34 + tw - 1 + 1
    const __bf16* inB = h3p + ((size_t)b * 64) * 1156
                            + (qy + 1 + py) * 34 + (qx + px);
    const v16bf* wf = (const v16bf*)wfrag + (size_t)cls * (8 * 2 * 32);

    const v8f zero8 = {0.f, 0.f, 0.f, 0.f, 0.f, 0.f, 0.f, 0.f};
    v8f acc[2] = {zero8, zero8};

#pragma unroll
    for (int kb = 0; kb < 256; kb += 32) {
        v16bf a;
#pragma unroll
        for (int v = 0; v < 8; ++v) {
#pragma unroll
            for (int j = 0; j < 2; ++j) {
                int k  = kb + lhalf * 16 + v * 2 + j;
                int ci = k >> 2, t = k & 3, th = t >> 1, tw = t & 1;
                // tw flipped in packing -> address ascends with k within pairs
                a[2 * v + j] = inB[ci * 1156 - th * 34 + tw];
            }
        }
        const int fb = (kb >> 5) * 2;
        v16bf bf0 = wf[((fb + 0) << 5) + lane];
        v16bf bf1 = wf[((fb + 1) << 5) + lane];
        acc[0] = __builtin_amdgcn_wmma_f32_16x16x32_bf16(false, a, false, bf0, (short)0, acc[0], false, false);
        acc[1] = __builtin_amdgcn_wmma_f32_16x16x32_bf16(false, a, false, bf1, (short)0, acc[1], false, false);
    }

#pragma unroll
    for (int nt = 0; nt < 2; ++nt) {
        int n = nt * 16 + lidx;
        float bv = bias[n];
#pragma unroll
        for (int r = 0; r < 8; ++r) {
            int mm  = mBase + lhalf * 8 + r;
            int bb  = mm >> 10;
            int pix = mm & 1023;
            int qyy = pix >> 5, qxx = pix & 31;
            float v = fmaxf(acc[nt][r] + bv, 0.0f);
            h4[(((size_t)bb * 32 + n) * 64 + 2 * qyy + py) * 64 + 2 * qxx + px] = (__bf16)v;
        }
    }
}

// ---------------------------------------------------------------------------
// VQ: nearest codebook entry per pixel (codebook cached in LDS), write z_q
// into padded zqp [128][64][34][34] interior (bf16) and accumulate
// vq_loss = mean((z - z_q)^2).
// ---------------------------------------------------------------------------
__global__ void __launch_bounds__(256) quantize_vq(
        const float* __restrict__ z, const float* __restrict__ codebook,
        __bf16* __restrict__ zqp, float* __restrict__ vq_out) {
    __shared__ __align__(16) float cb[128 * 64];
    __shared__ float cn[128];
    {   // vectorized LDS fill (2048 float4)
        const float4* c4 = (const float4*)codebook;
        float4* s4 = (float4*)cb;
        for (int i = threadIdx.x; i < 2048; i += 256) s4[i] = c4[i];
    }
    __syncthreads();
    if (threadIdx.x < 128) {
        float s = 0.0f;
#pragma unroll
        for (int j = 0; j < 64; ++j) { float c = cb[threadIdx.x * 64 + j]; s += c * c; }
        cn[threadIdx.x] = s;
    }
    __syncthreads();

    int m   = blockIdx.x * 256 + threadIdx.x;  // < 131072 pixels
    int b   = m >> 10;
    int rem = m & 1023;
    const float* zb = z + (size_t)b * 64 * 1024 + rem;   // channel stride 1024

    float zv[64];
#pragma unroll
    for (int j = 0; j < 64; ++j) zv[j] = zb[j * 1024];

    float best = 3.0e38f;
    int   bi   = 0;
    for (int e = 0; e < 128; ++e) {
        float dot = 0.0f;
#pragma unroll
        for (int j = 0; j < 64; ++j) dot += zv[j] * cb[e * 64 + j];
        float d = cn[e] - 2.0f * dot;           // |z|^2 constant for argmin
        if (d < best) { best = d; bi = e; }
    }

    int y = rem >> 5, x = rem & 31;
    __bf16* zqb = zqp + ((size_t)b * 64) * 1156 + (y + 1) * 34 + (x + 1);
    float vq = 0.0f;
#pragma unroll
    for (int j = 0; j < 64; ++j) {
        float c  = cb[bi * 64 + j];
        float df = zv[j] - c;
        vq += df * df;
        zqb[j * 1156] = (__bf16)c;
    }
#pragma unroll
    for (int off = 16; off > 0; off >>= 1) vq += __shfl_xor(vq, off, 32);
    if ((threadIdx.x & 31) == 0)
        atomicAdd(vq_out, vq * (1.0f / 8388608.0f));     // mean over 128*64*32*32
}

// ---------------------------------------------------------------------------
// deconv3: ConvTranspose2d 32->1, 4x4, stride 2, pad 1 + sigmoid, and
// recon_loss = mean((x_recon - x)^2) via wave reduction + atomics.
// ---------------------------------------------------------------------------
__global__ void __launch_bounds__(256) deconv3_sig(
        const __bf16* __restrict__ h4, const float* __restrict__ dw3 /*[32,1,4,4]*/,
        const float* __restrict__ db3, const float* __restrict__ x,
        float* __restrict__ out, float* __restrict__ recon_out) {
    int idx = blockIdx.x * 256 + threadIdx.x;          // (b*128+oy)*128+ox
    int ox = idx & 127;
    int oy = (idx >> 7) & 127;
    int b  = idx >> 14;
    const __bf16* inb = h4 + (size_t)b * 32 * 4096;    // [32][64][64]
    float acc = db3[0];
#pragma unroll
    for (int kh = 0; kh < 4; ++kh) {
        int ty = oy + 1 - kh;
        if (ty & 1) continue;
        int iy = ty >> 1;
        if ((unsigned)iy >= 64u) continue;
#pragma unroll
        for (int kw = 0; kw < 4; ++kw) {
            int tx = ox + 1 - kw;
            if (tx & 1) continue;
            int ix = tx >> 1;
            if ((unsigned)ix >= 64u) continue;
            const __bf16* ip = inb + iy * 64 + ix;
            const float*  wp = dw3 + kh * 4 + kw;
            for (int ci = 0; ci < 32; ++ci)
                acc += wp[ci * 16] * (float)ip[ci * 4096];
        }
    }
    float sig = 1.0f / (1.0f + expf(-acc));
    out[idx] = sig;
    float d  = sig - x[idx];
    float sq = d * d;
#pragma unroll
    for (int off = 16; off > 0; off >>= 1) sq += __shfl_xor(sq, off, 32);
    if ((threadIdx.x & 31) == 0)
        atomicAdd(recon_out, sq * (1.0f / 2097152.0f));
}

// ---------------------------------------------------------------------------
// Host launcher
// ---------------------------------------------------------------------------
extern "C" void kernel_launch(void* const* d_in, const int* in_sizes, int n_in,
                              void* d_out, int out_size, void* d_ws, size_t ws_size,
                              hipStream_t stream) {
    const float* x   = (const float*)d_in[0];
    const float* w1  = (const float*)d_in[1];
    const float* b1  = (const float*)d_in[2];
    const float* w2  = (const float*)d_in[3];
    const float* b2  = (const float*)d_in[4];
    const float* w3  = (const float*)d_in[5];
    const float* b3  = (const float*)d_in[6];
    const float* cb  = (const float*)d_in[7];
    const float* dw1 = (const float*)d_in[8];
    const float* db1 = (const float*)d_in[9];
    const float* dw2 = (const float*)d_in[10];
    const float* db2 = (const float*)d_in[11];
    const float* dw3 = (const float*)d_in[12];
    const float* db3 = (const float*)d_in[13];
    float* out = (float*)d_out;
    char*  ws  = (char*)d_ws;

    // workspace layout (bytes)
    __bf16* h1p  = (__bf16*)(ws + 0);                   // 35,684,352  [128,32,66,66]
    __bf16* h2p  = (__bf16*)(ws + (size_t)35684352);    // 18,939,904  [128,64,34,34]
    float*  z    = (float*) (ws + (size_t)54624256);    // 33,554,432  [128,64,32,32]
    __bf16* zqp  = (__bf16*)(ws + (size_t)88178688);    // 18,939,904  [128,64,34,34]
    __bf16* h3p  = (__bf16*)(ws + (size_t)107118592);   // 18,939,904  [128,64,34,34]
    __bf16* h4   = (__bf16*)(ws + (size_t)126058496);   // 33,554,432  [128,32,64,64]
    __bf16* w2p  = (__bf16*)(ws + (size_t)159612928);   //     65,536
    __bf16* w3p  = (__bf16*)(ws + (size_t)159678464);   //     73,728
    __bf16* dw1p = (__bf16*)(ws + (size_t)159752192);   //     73,728
    __bf16* dw2p = (__bf16*)(ws + (size_t)159825920);   //     65,536

    float* losses = out + 2097152;                      // [recon_loss, vq_loss]

    init_losses<<<1, 32, 0, stream>>>(losses);
    // pre-zero padded buffers so halos are 0 (interiors fully overwritten)
    zero_buf<<<8712, 256, 0, stream>>>((uint4*)h1p, 2230272);
    zero_buf<<<4624, 256, 0, stream>>>((uint4*)h2p, 1183744);
    zero_buf<<<4624, 256, 0, stream>>>((uint4*)zqp, 1183744);
    zero_buf<<<4624, 256, 0, stream>>>((uint4*)h3p, 1183744);
    pack_weights<<<144, 256, 0, stream>>>(w2, w3, dw1, dw2, w2p, w3p, dw1p, dw2p);
    conv1_relu<<<65536, 256, 0, stream>>>(x, w1, b1, h1p);
    // conv2: 32ch 4x4 s2, HP=66, ReLU, bf16 padded out
    conv_gemm<32, 4, 2, 66, true,  2><<<2048, 128, 0, stream>>>(h1p, w2p, b2, (void*)h2p);
    // conv3: 64ch 3x3 s1, HP=34, no ReLU, f32 flat out (z)
    conv_gemm<64, 3, 1, 34, false, 0><<<2048, 128, 0, stream>>>(h2p, w3p, b3, (void*)z);
    quantize_vq<<<512, 256, 0, stream>>>(z, cb, zqp, losses + 1);
    // deconv1 == 3x3 s1 conv with pre-flipped weights, ReLU, bf16 PADDED out
    conv_gemm<64, 3, 1, 34, true,  2><<<2048, 128, 0, stream>>>(zqp, dw1p, db1, (void*)h3p);
    // deconv2: 4 parity-class GEMMs on WMMA
    deconv2_gemm<<<dim3(2048, 4), 128, 0, stream>>>(h3p, dw2p, db2, h4);
    deconv3_sig<<<8192, 256, 0, stream>>>(h4, dw3, db3, x, out, losses);
}
